// Model_16982300688810
// MI455X (gfx1250) — matrix operand order repaired and checked by reading, not compiled, not run on hardware
//
#include <hip/hip_runtime.h>
#include <hip/hip_bf16.h>

// ---------------------------------------------------------------------------
// Model dims
// ---------------------------------------------------------------------------
#define Bc   4
#define TOc  48
#define TIc  96
#define Nc_  256
#define Dc   64
#define TDc  128
#define FFc  512
#define Hc   4
#define HBc  16
#define BNc  (Bc * Nc_)          // 1024

typedef __attribute__((ext_vector_type(16))) __bf16 v16bf;
typedef __attribute__((ext_vector_type(8)))  float  v8f;

__device__ __forceinline__ unsigned short f2bf(float f) {
    union { float f; unsigned u; } x; x.f = f;
    unsigned r = x.u + 0x7fffu + ((x.u >> 16) & 1u);
    return (unsigned short)(r >> 16);
}

struct BFrag { union { v16bf v; unsigned short u[16]; uint4 q[2]; }; };

// A-operand (16x32 bf16) from row-major [row][k] storage.
// lane (m = lane&15, hi = lane>>4) holds K = {k0+hi*16 .. +15}
// -> one contiguous 32-byte run (matches load_b_km).
__device__ __forceinline__ void load_a_rm(BFrag& f, const unsigned short* base,
                                          int ld, int row0, int k0, int lane) {
    int m = lane & 15, hi = lane >> 4;
    const unsigned short* p = base + (size_t)(row0 + m) * ld + k0 + hi * 16;
    f.q[0] = *(const uint4*)p;
    f.q[1] = *(const uint4*)(p + 8);
}

// B-operand (32x16 bf16) from K-major (transposed) storage Bt[n][k]:
// lane (nn = lane&15, hi = lane>>4) holds K = {k0+hi*16 .. +15} of column nn
// -> one contiguous 32-byte run.
__device__ __forceinline__ void load_b_km(BFrag& f, const unsigned short* base,
                                          int ldk, int k0, int col0, int lane) {
    int nn = lane & 15, hi = lane >> 4;
    const unsigned short* p = base + (size_t)(col0 + nn) * ldk + k0 + hi * 16;
    f.q[0] = *(const uint4*)p;
    f.q[1] = *(const uint4*)(p + 8);
}

__device__ __forceinline__ v8f wmma_bf16(const BFrag& a, const BFrag& b, v8f c) {
    return __builtin_amdgcn_wmma_f32_16x16x32_bf16(false, a.v, false, b.v,
                                                   (short)0, c, false, false);
}

__device__ __forceinline__ float wsum(float x) {
#pragma unroll
    for (int o = 16; o > 0; o >>= 1) x += __shfl_xor(x, o, 32);
    return x;
}
__device__ __forceinline__ float wmax32(float x) {
#pragma unroll
    for (int o = 16; o > 0; o >>= 1) x = fmaxf(x, __shfl_xor(x, o, 32));
    return x;
}

// ---------------------------------------------------------------------------
// Weight preprocessing: fp32 W[K][N] -> bf16 Wt[N][K]
// ---------------------------------------------------------------------------
__global__ __launch_bounds__(256) void convert_w_kernel(const float* __restrict__ W,
                                                        unsigned short* __restrict__ Wt,
                                                        int K, int N) {
    size_t idx = (size_t)blockIdx.x * 256 + threadIdx.x;
    int n = (int)(idx / K), k = (int)(idx % K);
    Wt[idx] = f2bf(W[(size_t)k * N + n]);
}

// ---------------------------------------------------------------------------
// (B,T,N,D) -> (B,N,T,D)
// ---------------------------------------------------------------------------
__global__ __launch_bounds__(256) void permute_btnd(const float* __restrict__ in,
                                                    float* __restrict__ out,
                                                    int T, int Nn, int D) {
    size_t idx = (size_t)blockIdx.x * 256 + threadIdx.x;
    int d = (int)(idx % D); size_t r = idx / D;
    int n = (int)(r % Nn);  r /= Nn;
    int t = (int)(r % T);   int b = (int)(r / T);
    out[(((size_t)b * Nn + n) * T + t) * D + d] = in[idx];
}

// ---------------------------------------------------------------------------
// Flat-token GEMM: C[M,Nc] = act(A[M,K] @ W + bias), W given bf16 K-major.
// block 256 (8 waves), tile 128x64, K-step 32.
// ---------------------------------------------------------------------------
template <int ACT>
__global__ __launch_bounds__(256) void gemm_kernel(const float* __restrict__ A,
                                                   const unsigned short* __restrict__ Wt,
                                                   const float* __restrict__ bias,
                                                   float* __restrict__ C,
                                                   int M, int K, int Nc) {
    __shared__ unsigned short sA[128 * 32];
    int tid = threadIdx.x, lane = tid & 31, wave = tid >> 5;
    int n_base = blockIdx.x * 64;
    size_t m_base = (size_t)blockIdx.y * 128;
    v8f acc[4] = {};
    for (int k0 = 0; k0 < K; k0 += 32) {
        __syncthreads();
        {   // stage A tile 128x32 (fp32 -> bf16)
            int m = tid >> 1, kh = (tid & 1) * 16;
            const float4* s4 = (const float4*)(A + (m_base + m) * (size_t)K + k0 + kh);
#pragma unroll
            for (int j = 0; j < 4; ++j) {
                float4 v = s4[j];
                unsigned short* d = &sA[m * 32 + kh + j * 4];
                d[0] = f2bf(v.x); d[1] = f2bf(v.y); d[2] = f2bf(v.z); d[3] = f2bf(v.w);
            }
        }
        __syncthreads();
        BFrag af; load_a_rm(af, sA, 32, wave * 16, 0, lane);
#pragma unroll
        for (int nt = 0; nt < 4; ++nt) {
            BFrag bf; load_b_km(bf, Wt, K, k0, n_base + nt * 16, lane);
            acc[nt] = wmma_bf16(af, bf, acc[nt]);
        }
    }
    int cc = lane & 15, hi = lane >> 4;
#pragma unroll
    for (int nt = 0; nt < 4; ++nt) {
        int col = n_base + nt * 16 + cc;
        float bb = bias[col];
#pragma unroll
        for (int j = 0; j < 8; ++j) {
            size_t row = m_base + wave * 16 + hi * 8 + j;
            float v = acc[nt][j] + bb;
            if (ACT == 1) v = v > 0.f ? v : 0.f;
            C[row * (size_t)Nc + col] = v;
        }
    }
}

// ---------------------------------------------------------------------------
// Small per-(b,n,head) attention. hd = 16, Tq = 48, TK in {48,96}.
// STORE_PROBS: write bf16 softmax probs (final cross layer) instead of V-apply.
// grid.x = BN*H, block 256 (8 waves -> 8 query rows at a time).
// ---------------------------------------------------------------------------
template <bool STORE_PROBS>
__global__ __launch_bounds__(256) void attn_kernel(const float* __restrict__ Q,
                                                   const float* __restrict__ Kb,
                                                   const float* __restrict__ Vb,
                                                   float* __restrict__ O,
                                                   unsigned short* __restrict__ P,
                                                   int TK) {
    __shared__ float sK[96 * 16];
    __shared__ float sV[96 * 16];
    __shared__ float sprob[8][96];
    int tid = threadIdx.x, lane = tid & 31, wave = tid >> 5;
    int h = blockIdx.x & 3, bn = blockIdx.x >> 2;
    for (int idx = tid; idx < TK * 16; idx += 256) {
        int t = idx >> 4, c = idx & 15;
        sK[idx] = Kb[((size_t)bn * TK + t) * 64 + h * 16 + c];
        if (!STORE_PROBS) sV[idx] = Vb[((size_t)bn * TK + t) * 64 + h * 16 + c];
    }
    __syncthreads();
    for (int row = wave; row < TOc; row += 8) {
        const float* qr = Q + ((size_t)bn * TOc + row) * 64 + h * 16;
        float qv[16];
#pragma unroll
        for (int c = 0; c < 16; ++c) qv[c] = qr[c];
        float sc[3];
#pragma unroll
        for (int j = 0; j < 3; ++j) {
            int t = lane + 32 * j;
            float s = -1e30f;
            if (t < TK) {
                float d = 0.f;
#pragma unroll
                for (int c = 0; c < 16; ++c) d += qv[c] * sK[t * 16 + c];
                s = d * 0.25f;   // 1/sqrt(16)
            }
            sc[j] = s;
        }
        float mx = wmax32(fmaxf(sc[0], fmaxf(sc[1], sc[2])));
        float lsum = 0.f;
#pragma unroll
        for (int j = 0; j < 3; ++j) {
            int t = lane + 32 * j;
            float p = 0.f;
            if (t < TK) { p = __expf(sc[j] - mx); sprob[wave][t] = p; }
            sc[j] = p; lsum += p;
        }
        float inv = 1.f / wsum(lsum);
        if constexpr (STORE_PROBS) {
            int b = bn >> 8, n = bn & 255;            // bn = b*256+n
            int a = h * 4 + b;
            unsigned short* dst = P + (((size_t)(a * 256 + n)) * TOc + row) * 96;
#pragma unroll
            for (int j = 0; j < 3; ++j) {
                int t = lane + 32 * j;
                if (t < TK) dst[t] = f2bf(sc[j] * inv);
            }
        } else {
            int c = lane & 15, half = lane >> 4;
            float a0 = 0.f;
            for (int t = half; t < TK; t += 2) a0 += sprob[wave][t] * sV[t * 16 + c];
            a0 += __shfl_down(a0, 16, 32);
            if (half == 0) O[((size_t)bn * TOc + row) * 64 + h * 16 + c] = a0 * inv;
        }
    }
}

// ---------------------------------------------------------------------------
// Out = LN(A + R), one wave per row, D in {64,128}
// ---------------------------------------------------------------------------
__global__ __launch_bounds__(256) void add_ln_kernel(const float* __restrict__ A,
                                                     const float* __restrict__ R,
                                                     const float* __restrict__ g,
                                                     const float* __restrict__ bta,
                                                     float* __restrict__ Out,
                                                     int rows, int D) {
    int wave = threadIdx.x >> 5, lane = threadIdx.x & 31;
    size_t row = (size_t)blockIdx.x * 8 + wave;
    if (row >= (size_t)rows) return;
    int nj = D >> 5;
    float x[4]; float s = 0.f;
    for (int j = 0; j < nj; ++j) {
        size_t idx = row * D + lane + 32 * j;
        x[j] = A[idx] + R[idx]; s += x[j];
    }
    s = wsum(s); float mu = s / D;
    float v = 0.f;
    for (int j = 0; j < nj; ++j) { float d = x[j] - mu; v += d * d; }
    v = wsum(v);
    float rstd = rsqrtf(v / D + 1e-5f);
    for (int j = 0; j < nj; ++j) {
        int c = lane + 32 * j;
        Out[row * D + c] = (x[j] - mu) * rstd * g[c] + bta[c];
    }
}

// ---------------------------------------------------------------------------
// Fused final cross layer. One workgroup per (n, b16, bq); grid (256,16,4).
// O_cat(48x128) = per-head A(48x96) @ Vslice(96x32); proj+LN; FF 128->512->128
// with residual + LN; resident in 84KB LDS; bf16 K-major weights streamed.
// ---------------------------------------------------------------------------
__global__ __launch_bounds__(256) void cross_fused_kernel(
    const unsigned short* __restrict__ attnP,  // (16,256,48,96) bf16
    const float* __restrict__ Vb,              // (BN,96,128) fp32
    const unsigned short* __restrict__ WoT,    // [128][128] bf16 (N-major)
    const float* __restrict__ bo,
    const float* __restrict__ lnG, const float* __restrict__ lnB,
    const unsigned short* __restrict__ W1T,    // [512][128]
    const float* __restrict__ b1,
    const unsigned short* __restrict__ W2T,    // [128][512]
    const float* __restrict__ b2,
    const float* __restrict__ ln2G, const float* __restrict__ ln2B,
    float* __restrict__ out) {
    __shared__ __align__(16) char smem[86016];
    unsigned short* sA  = (unsigned short*)smem;            // [4][48][96] bf16 (ph 0-1)
    unsigned short* sVt = (unsigned short*)(smem + 36864);  // [32][96] bf16 K-major (ph 0-1)
    unsigned short* sH  = (unsigned short*)smem;            // [48][512] bf16 (ph 4-5)
    unsigned short* sX  = (unsigned short*)(smem + 49152);  // [48][128] bf16 (O_cat, then t)
    float*          sP  = (float*)(smem + 61440);           // [48][128] fp32

    int tid = threadIdx.x, lane = tid & 31, wave = tid >> 5;
    int n = blockIdx.x, b16 = blockIdx.y, bq = blockIdx.z;
    int hv = b16 >> 2, bv = b16 & 3;

    // Phase 0: copy bf16 probs (uint-wide) + stage V slice transposed (K-major)
    for (int idx = tid; idx < 4 * 48 * 96 / 2; idx += 256) {
        int h = idx / 2304, rem = idx - h * 2304;
        ((unsigned*)sA)[idx] =
            ((const unsigned*)(attnP + ((size_t)((h * 4 + bq) * 256 + n)) * 4608))[rem];
    }
    {
        const float* vsrc = Vb + ((size_t)(bv * 256 + n) * 96) * 128 + hv * 32;
        for (int idx = tid; idx < 96 * 32; idx += 256) {
            int t = idx >> 5, c = idx & 31;
            sVt[c * 96 + t] = f2bf(vsrc[(size_t)t * 128 + c]);
        }
    }
    __syncthreads();

    // Phase 1: O_cat = A_h @ Vslice  -> sX (48x128 bf16). Wave w owns cols [16w,16w+16).
    {
        int nt = wave, h = nt >> 1, vcol0 = (nt & 1) * 16;
        const unsigned short* Ah = sA + h * 48 * 96;
        v8f acc[3] = {};
        for (int k0 = 0; k0 < 96; k0 += 32) {
            BFrag bf; load_b_km(bf, sVt, 96, k0, vcol0, lane);
#pragma unroll
            for (int mt = 0; mt < 3; ++mt) {
                BFrag af; load_a_rm(af, Ah, 96, mt * 16, k0, lane);
                acc[mt] = wmma_bf16(af, bf, acc[mt]);
            }
        }
        int cc = lane & 15, hi = lane >> 4;
#pragma unroll
        for (int mt = 0; mt < 3; ++mt)
#pragma unroll
            for (int j = 0; j < 8; ++j) {
                int row = mt * 16 + hi * 8 + j;
                sX[row * 128 + nt * 16 + cc] = f2bf(acc[mt][j]);
            }
    }
    __syncthreads();

    // Phase 2: proj = O_cat @ Wo + bo -> sP fp32
    {
        int col0 = wave * 16;
        v8f acc[3] = {};
        for (int k0 = 0; k0 < 128; k0 += 32) {
            BFrag bf; load_b_km(bf, WoT, 128, k0, col0, lane);
#pragma unroll
            for (int mt = 0; mt < 3; ++mt) {
                BFrag af; load_a_rm(af, sX, 128, mt * 16, k0, lane);
                acc[mt] = wmma_bf16(af, bf, acc[mt]);
            }
        }
        int cc = lane & 15, hi = lane >> 4;
        float bb = bo[col0 + cc];
#pragma unroll
        for (int mt = 0; mt < 3; ++mt)
#pragma unroll
            for (int j = 0; j < 8; ++j) {
                int row = mt * 16 + hi * 8 + j;
                sP[row * 128 + col0 + cc] = acc[mt][j] + bb;
            }
    }
    __syncthreads();

    // Phase 3: t = LN(proj) -> sP fp32 (residual) + sX bf16 (GEMM operand)
    for (int row = wave; row < 48; row += 8) {
        float x[4]; float s = 0.f;
#pragma unroll
        for (int j = 0; j < 4; ++j) { x[j] = sP[row * 128 + lane + 32 * j]; s += x[j]; }
        s = wsum(s); float mu = s * (1.f / 128.f);
        float v = 0.f;
#pragma unroll
        for (int j = 0; j < 4; ++j) { float d = x[j] - mu; v += d * d; }
        v = wsum(v);
        float rstd = rsqrtf(v * (1.f / 128.f) + 1e-5f);
#pragma unroll
        for (int j = 0; j < 4; ++j) {
            int c = lane + 32 * j;
            float t = (x[j] - mu) * rstd * lnG[c] + lnB[c];
            sP[row * 128 + c] = t;
            sX[row * 128 + c] = f2bf(t);
        }
    }
    __syncthreads();

    // Phase 4: H = relu(t @ W1 + b1) -> sH (48x512 bf16). A-frags cached.
    {
        BFrag af[3][4];
#pragma unroll
        for (int mt = 0; mt < 3; ++mt)
#pragma unroll
            for (int kt = 0; kt < 4; ++kt) load_a_rm(af[mt][kt], sX, 128, mt * 16, kt * 32, lane);
#pragma unroll
        for (int jt = 0; jt < 4; ++jt) {
            int col0 = (wave * 4 + jt) * 16;
            v8f acc[3] = {};
#pragma unroll
            for (int kt = 0; kt < 4; ++kt) {
                BFrag bf; load_b_km(bf, W1T, 128, kt * 32, col0, lane);
#pragma unroll
                for (int mt = 0; mt < 3; ++mt) acc[mt] = wmma_bf16(af[mt][kt], bf, acc[mt]);
            }
            int cc = lane & 15, hi = lane >> 4;
            float bb = b1[col0 + cc];
#pragma unroll
            for (int mt = 0; mt < 3; ++mt)
#pragma unroll
                for (int j = 0; j < 8; ++j) {
                    int row = mt * 16 + hi * 8 + j;
                    float v = acc[mt][j] + bb;
                    sH[row * 512 + col0 + cc] = f2bf(v > 0.f ? v : 0.f);
                }
        }
    }
    __syncthreads();

    // Phase 5: F = H @ W2 + b2; sP = t + F
    {
        int col0 = wave * 16;
        v8f acc[3] = {};
        for (int kt = 0; kt < 16; ++kt) {
            BFrag bf; load_b_km(bf, W2T, 512, kt * 32, col0, lane);
#pragma unroll
            for (int mt = 0; mt < 3; ++mt) {
                BFrag af; load_a_rm(af, sH, 512, mt * 16, kt * 32, lane);
                acc[mt] = wmma_bf16(af, bf, acc[mt]);
            }
        }
        int cc = lane & 15, hi = lane >> 4;
        float bb = b2[col0 + cc];
#pragma unroll
        for (int mt = 0; mt < 3; ++mt)
#pragma unroll
            for (int j = 0; j < 8; ++j) {
                int row = mt * 16 + hi * 8 + j;
                sP[row * 128 + col0 + cc] += acc[mt][j] + bb;
            }
    }
    __syncthreads();

    // Phase 6: out = LN(sP) -> d_out at (bq, row, n, b16, c)
    for (int row = wave; row < 48; row += 8) {
        float x[4]; float s = 0.f;
#pragma unroll
        for (int j = 0; j < 4; ++j) { x[j] = sP[row * 128 + lane + 32 * j]; s += x[j]; }
        s = wsum(s); float mu = s * (1.f / 128.f);
        float v = 0.f;
#pragma unroll
        for (int j = 0; j < 4; ++j) { float d = x[j] - mu; v += d * d; }
        v = wsum(v);
        float rstd = rsqrtf(v * (1.f / 128.f) + 1e-5f);
        float* dst = out + (((size_t)(bq * 48 + row) * 256 + n) * 16 + b16) * 128;
#pragma unroll
        for (int j = 0; j < 4; ++j) {
            int c = lane + 32 * j;
            dst[c] = (x[j] - mu) * rstd * ln2G[c] + ln2B[c];
        }
    }
}

// ---------------------------------------------------------------------------
// Host orchestration
// ---------------------------------------------------------------------------
extern "C" void kernel_launch(void* const* d_in, const int* in_sizes, int n_in,
                              void* d_out, int out_size, void* d_ws, size_t ws_size,
                              hipStream_t stream) {
    (void)in_sizes; (void)n_in; (void)out_size; (void)ws_size;
    auto IN = [&](int i) { return (const float*)d_in[i]; };

    // Workspace carve (fp32 activations, then bf16 regions)
    float* ws = (float*)d_ws;
    float* yb  = ws;                    // 1024*48*64
    float* xb  = yb  + 3145728;         // 1024*96*64
    float* xgb = xb  + 6291456;         // 1024*96*128
    float* qb  = xgb + 12582912;        // 1024*48*64
    float* kb  = qb  + 3145728;         // 1024*96*64
    float* vb  = kb  + 6291456;         // 1024*96*128
    float* ob  = vb  + 12582912;        // 1024*48*64
    float* p1  = ob  + 3145728;         // 1024*48*64
    float* tb  = p1  + 3145728;         // 1024*48*64
    float* ffb = tb  + 3145728;         // 1024*48*512
    unsigned short* attnW = (unsigned short*)(ffb + 25165824);  // 16*256*48*96 bf16
    unsigned short* wbf   = attnW + 18874368;                   // bf16 weights

    // --- Pre-convert all weights to bf16 K-major (Wt[N][K]) ---
    size_t woff = 0;
    auto cvt = [&](int leaf, int K, int N) -> const unsigned short* {
        unsigned short* dst = wbf + woff;
        woff += (size_t)K * N;
        convert_w_kernel<<<(K * N) / 256, 256, 0, stream>>>(IN(leaf), dst, K, N);
        return dst;
    };
    struct LayerW { const unsigned short *q, *k, *v, *o, *w1, *w2; };
    // params leaf map (jax pytree, sorted dict keys): 3..18 cross, 19..50 crosst, 51..82 self
    LayerW LW[4]; int bases[4] = {51, 19, 67, 35};  // self0, crosst0, self1, crosst1
    for (int l = 0; l < 4; ++l) {
        int base = bases[l];
        LW[l].q  = cvt(base + 8, 64, 64);
        LW[l].k  = cvt(base + 4, 64, 64);
        LW[l].v  = cvt(base + 10, 64, 64);
        LW[l].o  = cvt(base + 6, 64, 64);
        LW[l].w1 = cvt(base + 0, 64, 512);
        LW[l].w2 = cvt(base + 2, 512, 64);
    }
    const unsigned short* cQ  = cvt(7, 64, 64);
    const unsigned short* cK  = cvt(3, 64, 64);
    const unsigned short* cV  = cvt(9, 128, 128);
    const unsigned short* cO  = cvt(5, 128, 128);
    const unsigned short* cW1 = cvt(11, 128, 512);
    const unsigned short* cW2 = cvt(13, 512, 128);

    // Transpose inputs (B,T,N,D)->(B,N,T,D)
    permute_btnd<<<12288, 256, 0, stream>>>(IN(0), yb, TOc, Nc_, Dc);
    permute_btnd<<<24576, 256, 0, stream>>>(IN(1), xb, TIc, Nc_, Dc);
    permute_btnd<<<49152, 256, 0, stream>>>(IN(2), xgb, TIc, Nc_, TDc);

    const int Mq = BNc * TOc;           // 49152
    auto run_self = [&](const float* kvsrc, int TK, int base, const LayerW& W) {
        int Mkv = BNc * TK;
        gemm_kernel<0><<<dim3(1, Mq / 128), 256, 0, stream>>>(yb, W.q, IN(base + 9), qb, Mq, 64, 64);
        gemm_kernel<0><<<dim3(1, Mkv / 128), 256, 0, stream>>>(kvsrc, W.k, IN(base + 5), kb, Mkv, 64, 64);
        gemm_kernel<0><<<dim3(1, Mkv / 128), 256, 0, stream>>>(kvsrc, W.v, IN(base + 11), vb, Mkv, 64, 64);
        attn_kernel<false><<<BNc * Hc, 256, 0, stream>>>(qb, kb, vb, ob, nullptr, TK);
        gemm_kernel<0><<<dim3(1, Mq / 128), 256, 0, stream>>>(ob, W.o, IN(base + 7), p1, Mq, 64, 64);
        add_ln_kernel<<<Mq / 8, 256, 0, stream>>>(yb, p1, IN(base + 13), IN(base + 12), tb, Mq, 64);
        gemm_kernel<1><<<dim3(8, Mq / 128), 256, 0, stream>>>(tb, W.w1, IN(base + 1), ffb, Mq, 64, 512);
        gemm_kernel<0><<<dim3(1, Mq / 128), 256, 0, stream>>>(ffb, W.w2, IN(base + 3), p1, Mq, 512, 64);
        add_ln_kernel<<<Mq / 8, 256, 0, stream>>>(tb, p1, IN(base + 15), IN(base + 14), yb, Mq, 64);
    };

    for (int i = 0; i < 2; ++i) {
        run_self(yb, TOc, 51 + 16 * i, LW[2 * i]);      // self layer i  (K/V = y)
        run_self(xb, TIc, 19 + 16 * i, LW[2 * i + 1]);  // crosst layer i (K/V = x_time)
    }

    // Final cross layer
    const int Mkv = BNc * TIc;          // 98304
    gemm_kernel<0><<<dim3(1, Mq / 128), 256, 0, stream>>>(yb, cQ, IN(8), qb, Mq, 64, 64);
    gemm_kernel<0><<<dim3(1, Mkv / 128), 256, 0, stream>>>(xb, cK, IN(4), kb, Mkv, 64, 64);
    gemm_kernel<0><<<dim3(2, Mkv / 128), 256, 0, stream>>>(xgb, cV, IN(10), vb, Mkv, 128, 128);
    attn_kernel<true><<<BNc * Hc, 256, 0, stream>>>(qb, kb, nullptr, nullptr, attnW, TIc);
    cross_fused_kernel<<<dim3(Nc_, HBc, Bc), 256, 0, stream>>>(
        attnW, vb, cO, IN(6), IN(18), IN(17), cW1, IN(12), cW2, IN(14),
        IN(16), IN(15), (float*)d_out);
}
